// ConvTransformerEncoderLayer_22539988369989
// MI455X (gfx1250) — compile-verified, matrix operand order checked by reading, not run on hardware
//
#include <hip/hip_runtime.h>
#include <hip/hip_bf16.h>

#define D_MODEL 512
#define NHEAD   8
#define HEAD_DIM 64
#define DFF     2048
#define SEQ     1024
#define BATCH   8
#define ROWS    (BATCH * SEQ)       // 8192
#define KC      (3 * D_MODEL)       // 1536 im2col K for QK convs

typedef __attribute__((ext_vector_type(16))) __bf16 bf16x16;
typedef __attribute__((ext_vector_type(8)))  __bf16 bf16x8;
typedef __attribute__((ext_vector_type(8)))  float  v8f;

union Frag { bf16x16 v; bf16x8 h[2]; };

// ---------------------------------------------------------------------------
// Packing / conversion kernels
// ---------------------------------------------------------------------------

// Wq/Wk: [O=512, I=512, K=3] f32  ->  Wt[o, k*512 + i] bf16  (i.e. [N, K] for GEMM)
__global__ __launch_bounds__(256) void pack_qk_w(const float* __restrict__ w,
                                                 __bf16* __restrict__ wt) {
    int idx = blockIdx.x * 256 + threadIdx.x;          // 512*1536 total
    int o = idx / KC, c = idx % KC;
    int k = c / D_MODEL, i = c % D_MODEL;
    wt[idx] = (__bf16)w[(size_t)o * KC + i * 3 + k];
}

__global__ __launch_bounds__(256) void f32_to_bf16(const float* __restrict__ in,
                                                   __bf16* __restrict__ out, int n) {
    int i = blockIdx.x * 256 + threadIdx.x;
    if (i < n) out[i] = (__bf16)in[i];
}

// src [B, S, 512] f32 -> Xc [B*S, 1536] bf16 with zero pad at sequence edges
__global__ __launch_bounds__(256) void im2col_bf16(const float* __restrict__ src,
                                                   __bf16* __restrict__ xc) {
    size_t idx = (size_t)blockIdx.x * 256 + threadIdx.x;   // 8192*1536 total
    int c = (int)(idx % KC);
    size_t row = idx / KC;
    int k = c / D_MODEL, i = c % D_MODEL;
    int s = (int)(row % SEQ);
    int b = (int)(row / SEQ);
    int ss = s + k - 1;
    float v = (ss >= 0 && ss < SEQ) ? src[((size_t)b * SEQ + ss) * D_MODEL + i] : 0.f;
    xc[idx] = (__bf16)v;
}

// ---------------------------------------------------------------------------
// Tiled bf16 WMMA GEMM:  C[M,N] = scale * (A[M,K] @ op(B)) + bias, opt ReLU.
//   BT=true : B given as [N,K] row-major (X @ W.T with W as stored); tiles are
//             staged with GLOBAL_LOAD_ASYNC_TO_LDS_B128 into DOUBLE-BUFFERED
//             LDS, next-tile DMA overlapped with current-tile WMMA (ASYNCcnt
//             pipelining: issue group k+1, s_wait_asynccnt<=3 for group k).
//   BT=false: B given as [K,N] row-major (B staged via VGPR transpose,
//             A still via async DMA; single buffer).
// Block tile 128x64, BK=32; 8 waves (wave32), each wave = 2x2 WMMA 16x16 tiles.
// LDS buffer layout (single block so async byte offsets match pointer reads):
//   As: bytes [0, 8192)      = [128][32] bf16, row stride 64 B
//   Bs: bytes [8192, 12288)  = [64][32]  bf16 (N-major), row stride 64 B
//   second buffer at +12288
// ---------------------------------------------------------------------------
#define AS_BYTES  (128 * 32 * 2)
#define BUF_BYTES (128 * 32 * 2 + 64 * 32 * 2)   // 12288
#define BUF_HALVES (BUF_BYTES / 2)

template <bool BT, bool BIAS, bool RELU, bool WF32, bool WB16>
__global__ __launch_bounds__(256)
void gemm_bf16(const __bf16* __restrict__ A, int lda,
               const __bf16* __restrict__ B, int ldb,
               float* __restrict__ Cf, __bf16* __restrict__ Cb, int ldc,
               const float* __restrict__ bias, float scale,
               int M, int N, int K) {
    __shared__ __bf16 smem[2 * BUF_HALVES];      // 24576 B, two buffers

    const int tid  = threadIdx.x;
    const int bm   = blockIdx.y * 128;
    const int bn   = blockIdx.x * 64;
    const int w    = tid >> 5;
    const int lane = tid & 31;
    const int wm   = (w & 3) * 32;
    const int wn   = (w >> 2) * 32;
    const int half = lane >> 4;
    const int lr   = lane & 15;
    const int ks   = half * 8;       // per-lane K start per ISA 16-bit A layout

    // staging assignments
    const int arow = tid >> 1, aseg = tid & 1;       // A: 128 rows x 2 x 16 halves
    const int brow = tid >> 2, bseg = tid & 3;       // B: 64 rows x 4 x 8 halves
    const unsigned aLds = (unsigned)(arow * 64 + aseg * 32);            // bytes
    const unsigned bLds = (unsigned)(AS_BYTES + brow * 64 + bseg * 16); // bytes
    const __bf16* agBase = &A[(size_t)(bm + arow) * lda + aseg * 16];
    const __bf16* bgBase = BT ? &B[(size_t)(bn + brow) * ldb + bseg * 8] : A;

    // issue one 3-load async group for K-chunk kk into LDS buffer at `base`
    auto issue = [&](int kk, unsigned base) {
        const __bf16* ag = agBase + kk;
        asm volatile("global_load_async_to_lds_b128 %0, %1, off"
                     :: "v"(aLds + base), "v"(ag) : "memory");
        asm volatile("global_load_async_to_lds_b128 %0, %1, off"
                     :: "v"(aLds + base + 16u), "v"(ag + 8) : "memory");
        if (BT) {
            const __bf16* bg = bgBase + kk;
            asm volatile("global_load_async_to_lds_b128 %0, %1, off"
                         :: "v"(bLds + base), "v"(bg) : "memory");
        }
    };

    v8f acc[2][2] = {};
    if (BT) issue(0, 0);

    for (int k0 = 0; k0 < K; k0 += 32) {
        unsigned cbase = ((unsigned)(k0 >> 5) & 1u) * BUF_BYTES;
        if (BT) {
            if (k0 + 32 < K) {
                issue(k0 + 32, cbase ^ BUF_BYTES);   // overlap next DMA w/ compute
                asm volatile("s_wait_asynccnt 0x3" ::: "memory"); // group k done
            } else {
                asm volatile("s_wait_asynccnt 0x0" ::: "memory");
            }
        } else {
            cbase = 0;
            issue(k0, 0);                            // A tile via async DMA
#pragma unroll
            for (int i = 0; i < 8; ++i) {            // transpose [K,N] -> Bs[n][k]
                int idx = tid + i * 256;             // 32*64 elements
                int k = idx >> 6, n = idx & 63;
                smem[AS_BYTES / 2 + n * 32 + k] = B[(size_t)(k0 + k) * ldb + bn + n];
            }
            asm volatile("s_wait_asynccnt 0x0" ::: "memory");
        }
        __syncthreads();

        // ---- fragments from LDS (two b128 loads each, ISA lane layout) ----
        const __bf16* sA = smem + (cbase >> 1);
        const __bf16* sB = sA + AS_BYTES / 2;
        Frag fa[2], fb[2];
#pragma unroll
        for (int i = 0; i < 2; ++i) {
            int row = wm + i * 16 + lr;
            fa[i].h[0] = *(const bf16x8*)(sA + row * 32 + ks);
            fa[i].h[1] = *(const bf16x8*)(sA + row * 32 + ks + 16);
        }
#pragma unroll
        for (int j = 0; j < 2; ++j) {
            int n = wn + j * 16 + lr;
            fb[j].h[0] = *(const bf16x8*)(sB + n * 32 + ks);
            fb[j].h[1] = *(const bf16x8*)(sB + n * 32 + ks + 16);
        }
#pragma unroll
        for (int i = 0; i < 2; ++i)
#pragma unroll
            for (int j = 0; j < 2; ++j)
                acc[i][j] = __builtin_amdgcn_wmma_f32_16x16x32_bf16(
                    false, fa[i].v, false, fb[j].v, (short)0, acc[i][j],
                    false, false);
        __syncthreads();
    }

    // ---- epilogue (f32 C layout: VGPR r -> M = r + half*8, N = lr) ----
#pragma unroll
    for (int i = 0; i < 2; ++i) {
#pragma unroll
        for (int j = 0; j < 2; ++j) {
            int col = bn + wn + j * 16 + lr;
            float bv = BIAS ? bias[col] : 0.f;
#pragma unroll
            for (int r = 0; r < 8; ++r) {
                int row = bm + wm + i * 16 + r + half * 8;
                float v = acc[i][j][r] * scale + bv;
                if (RELU) v = v > 0.f ? v : 0.f;
                size_t off = (size_t)row * ldc + col;
                if (WF32) Cf[off] = v;
                if (WB16) Cb[off] = (__bf16)v;
            }
        }
    }
}

// ---------------------------------------------------------------------------
// Row softmax over 1024 cols, f32 in -> bf16 out. One block per row.
// ---------------------------------------------------------------------------
__global__ __launch_bounds__(256) void softmax_row(const float* __restrict__ in,
                                                   __bf16* __restrict__ out) {
    __shared__ float red[256];
    const int row = blockIdx.x, tid = threadIdx.x;
    const float* p = in + (size_t)row * SEQ;
    float v[4], m = -3.4e38f;
#pragma unroll
    for (int i = 0; i < 4; ++i) { v[i] = p[tid + i * 256]; m = fmaxf(m, v[i]); }
    red[tid] = m; __syncthreads();
    for (int s = 128; s > 0; s >>= 1) {
        if (tid < s) red[tid] = fmaxf(red[tid], red[tid + s]);
        __syncthreads();
    }
    m = red[0]; __syncthreads();
    float sum = 0.f;
#pragma unroll
    for (int i = 0; i < 4; ++i) { v[i] = __expf(v[i] - m); sum += v[i]; }
    red[tid] = sum; __syncthreads();
    for (int s = 128; s > 0; s >>= 1) {
        if (tid < s) red[tid] += red[tid + s];
        __syncthreads();
    }
    float inv = 1.f / red[0];
    __bf16* q = out + (size_t)row * SEQ;
#pragma unroll
    for (int i = 0; i < 4; ++i) q[tid + i * 256] = (__bf16)(v[i] * inv);
}

// ---------------------------------------------------------------------------
// y = LN(a + b) * g + be ; 512 cols per row, one block per row.
// ---------------------------------------------------------------------------
template <bool WB16>
__global__ __launch_bounds__(256)
void layernorm_res(const float* __restrict__ a, const float* __restrict__ b,
                   const float* __restrict__ g, const float* __restrict__ be,
                   float* __restrict__ of, __bf16* __restrict__ ob) {
    __shared__ float red[256];
    const int row = blockIdx.x, tid = threadIdx.x;
    const size_t base = (size_t)row * D_MODEL;
    float x0 = a[base + tid] + b[base + tid];
    float x1 = a[base + tid + 256] + b[base + tid + 256];
    red[tid] = x0 + x1; __syncthreads();
    for (int s = 128; s > 0; s >>= 1) {
        if (tid < s) red[tid] += red[tid + s];
        __syncthreads();
    }
    float mu = red[0] * (1.f / D_MODEL); __syncthreads();
    float d0 = x0 - mu, d1 = x1 - mu;
    red[tid] = d0 * d0 + d1 * d1; __syncthreads();
    for (int s = 128; s > 0; s >>= 1) {
        if (tid < s) red[tid] += red[tid + s];
        __syncthreads();
    }
    float rstd = rsqrtf(red[0] * (1.f / D_MODEL) + 1e-5f);
    float y0 = d0 * rstd * g[tid] + be[tid];
    float y1 = d1 * rstd * g[tid + 256] + be[tid + 256];
    of[base + tid] = y0; of[base + tid + 256] = y1;
    if (WB16) { ob[base + tid] = (__bf16)y0; ob[base + tid + 256] = (__bf16)y1; }
}

// ---------------------------------------------------------------------------
extern "C" void kernel_launch(void* const* d_in, const int* in_sizes, int n_in,
                              void* d_out, int out_size, void* d_ws, size_t ws_size,
                              hipStream_t stream) {
    const float* src = (const float*)d_in[0];
    const float* Wq  = (const float*)d_in[1];
    const float* bq  = (const float*)d_in[2];
    const float* Wk  = (const float*)d_in[3];
    const float* bk  = (const float*)d_in[4];
    const float* Wv  = (const float*)d_in[5];
    const float* bv  = (const float*)d_in[6];
    const float* Wo  = (const float*)d_in[7];
    const float* bo  = (const float*)d_in[8];
    const float* W1  = (const float*)d_in[9];
    const float* b1  = (const float*)d_in[10];
    const float* W2  = (const float*)d_in[11];
    const float* b2  = (const float*)d_in[12];
    const float* g1  = (const float*)d_in[13];
    const float* be1 = (const float*)d_in[14];
    const float* g2  = (const float*)d_in[15];
    const float* be2 = (const float*)d_in[16];
    float* out = (float*)d_out;

    // ---- workspace layout ----
    char* p = (char*)d_ws;
    auto take = [&](size_t bytes) {
        char* r = p;
        p += (bytes + 255) & ~(size_t)255;
        return r;
    };
    __bf16* Xc   = (__bf16*)take((size_t)ROWS * KC * 2);        // 25.2 MB
    __bf16* Qb   = (__bf16*)take((size_t)ROWS * D_MODEL * 2);   // 8.4 MB
    __bf16* Kb   = (__bf16*)take((size_t)ROWS * D_MODEL * 2);
    __bf16* Vb   = (__bf16*)take((size_t)ROWS * D_MODEL * 2);
    __bf16* AVb  = (__bf16*)take((size_t)ROWS * D_MODEL * 2);
    float*  scoF = (float*) take((size_t)SEQ * SEQ * 4);        // 4 MB (per b,h)
    __bf16* attB = (__bf16*)take((size_t)SEQ * SEQ * 2);        // 2 MB
    float*  saF  = (float*) take((size_t)ROWS * D_MODEL * 4);
    float*  x1F  = (float*) take((size_t)ROWS * D_MODEL * 4);
    __bf16* x1B  = (__bf16*)take((size_t)ROWS * D_MODEL * 2);
    __bf16* ff1B = (__bf16*)take((size_t)ROWS * DFF * 2);       // 33.6 MB
    float*  ff2F = (float*) take((size_t)ROWS * D_MODEL * 4);
    __bf16* Wqt  = (__bf16*)take((size_t)D_MODEL * KC * 2);
    __bf16* Wkt  = (__bf16*)take((size_t)D_MODEL * KC * 2);
    __bf16* Wvb  = (__bf16*)take((size_t)D_MODEL * D_MODEL * 2);
    __bf16* Wob  = (__bf16*)take((size_t)D_MODEL * D_MODEL * 2);
    __bf16* W1b  = (__bf16*)take((size_t)DFF * D_MODEL * 2);
    __bf16* W2b  = (__bf16*)take((size_t)D_MODEL * DFF * 2);
    (void)ws_size; (void)in_sizes; (void)n_in; (void)out_size;

    // ---- weight packing ----
    pack_qk_w<<<(D_MODEL * KC) / 256, 256, 0, stream>>>(Wq, Wqt);
    pack_qk_w<<<(D_MODEL * KC) / 256, 256, 0, stream>>>(Wk, Wkt);
    f32_to_bf16<<<(D_MODEL * D_MODEL + 255) / 256, 256, 0, stream>>>(Wv, Wvb, D_MODEL * D_MODEL);
    f32_to_bf16<<<(D_MODEL * D_MODEL + 255) / 256, 256, 0, stream>>>(Wo, Wob, D_MODEL * D_MODEL);
    f32_to_bf16<<<(DFF * D_MODEL + 255) / 256, 256, 0, stream>>>(W1, W1b, DFF * D_MODEL);
    f32_to_bf16<<<(D_MODEL * DFF + 255) / 256, 256, 0, stream>>>(W2, W2b, D_MODEL * DFF);
    im2col_bf16<<<(ROWS * KC) / 256, 256, 0, stream>>>(src, Xc);

    auto ggrid = [](int M, int N) { return dim3(N / 64, M / 128, 1); };

    // ---- QKV projections (conv as GEMM) ----
    gemm_bf16<true, true, false, false, true><<<ggrid(ROWS, D_MODEL), 256, 0, stream>>>(
        Xc, KC, Wqt, KC, nullptr, Qb, D_MODEL, bq, 1.f, ROWS, D_MODEL, KC);
    gemm_bf16<true, true, false, false, true><<<ggrid(ROWS, D_MODEL), 256, 0, stream>>>(
        Xc, KC, Wkt, KC, nullptr, Kb, D_MODEL, bk, 1.f, ROWS, D_MODEL, KC);
    gemm_bf16<true, true, false, false, true><<<ggrid(ROWS, D_MODEL), 256, 0, stream>>>(
        Xc + D_MODEL, KC, Wvb, D_MODEL, nullptr, Vb, D_MODEL, bv, 1.f, ROWS, D_MODEL, D_MODEL);

    // ---- attention, per (batch, head) ----
    for (int z = 0; z < BATCH * NHEAD; ++z) {
        int b = z >> 3, h = z & 7;
        size_t hoff = (size_t)b * SEQ * D_MODEL + (size_t)h * HEAD_DIM;
        // scores = Qh @ Kh.T / 64   [1024 x 1024], K = 64
        gemm_bf16<true, false, false, true, false><<<ggrid(SEQ, SEQ), 256, 0, stream>>>(
            Qb + hoff, D_MODEL, Kb + hoff, D_MODEL, scoF, nullptr, SEQ,
            nullptr, 1.f / HEAD_DIM, SEQ, SEQ, HEAD_DIM);
        softmax_row<<<SEQ, 256, 0, stream>>>(scoF, attB);
        // AV = att @ Vh   [1024 x 64], K = 1024 ; B is [K,N] (non-transposed)
        gemm_bf16<false, false, false, false, true><<<ggrid(SEQ, HEAD_DIM), 256, 0, stream>>>(
            attB, SEQ, Vb + hoff, D_MODEL, nullptr, AVb + hoff, D_MODEL,
            nullptr, 1.f, SEQ, HEAD_DIM, SEQ);
    }

    // ---- output projection + LN1 ----
    gemm_bf16<true, true, false, true, false><<<ggrid(ROWS, D_MODEL), 256, 0, stream>>>(
        AVb, D_MODEL, Wob, D_MODEL, saF, nullptr, D_MODEL, bo, 1.f, ROWS, D_MODEL, D_MODEL);
    layernorm_res<true><<<ROWS, 256, 0, stream>>>(src, saF, g1, be1, x1F, x1B);

    // ---- FFN + LN2 ----
    gemm_bf16<true, true, true, false, true><<<ggrid(ROWS, DFF), 256, 0, stream>>>(
        x1B, D_MODEL, W1b, D_MODEL, nullptr, ff1B, DFF, b1, 1.f, ROWS, DFF, D_MODEL);
    gemm_bf16<true, true, false, true, false><<<ggrid(ROWS, D_MODEL), 256, 0, stream>>>(
        ff1B, DFF, W2b, DFF, ff2F, nullptr, D_MODEL, b2, 1.f, ROWS, D_MODEL, DFF);
    layernorm_res<false><<<ROWS, 256, 0, stream>>>(x1F, ff2F, g2, be2, out, nullptr);
}
